// BLinear_35742717838030
// MI455X (gfx1250) — matrix operand order repaired and checked by reading, not compiled, not run on hardware
//
#include <hip/hip_runtime.h>
#include <hip/hip_bf16.h>

typedef __attribute__((ext_vector_type(16))) __bf16 v16bf;
typedef __attribute__((ext_vector_type(8)))  __bf16 v8bf;
typedef __attribute__((ext_vector_type(4)))  __bf16 v4bf;
typedef __attribute__((ext_vector_type(8)))  float  v8f;
typedef __attribute__((ext_vector_type(4)))  float  v4f;
typedef __attribute__((ext_vector_type(4)))  int    v4i;

#define N_IN   1024
#define N_OUT  1024
#define BATCH  4096
#define BK     32       // K step == WMMA K
#define LDSS   40       // padded LDS row stride (halves): 20 dwords -> conflict-free frag reads

// ---------------- gfx1250 async global->LDS support (guarded) ----------------
#if defined(__has_builtin)
#if __has_builtin(__builtin_amdgcn_global_load_async_to_lds_b128)
#define HAVE_ASYNC_LDS 1
#endif
#endif

#ifdef HAVE_ASYNC_LDS
#if defined(__has_builtin)
#if __has_builtin(__builtin_amdgcn_s_wait_asynccnt)
#define ASYNC_WAIT(n) __builtin_amdgcn_s_wait_asynccnt(n)
#endif
#endif
#ifndef ASYNC_WAIT
#define ASYNC_WAIT(n) asm volatile("s_wait_asynccnt %0" :: "n"(n) : "memory")
#endif
#else
#define ASYNC_WAIT(n)
#endif

__device__ __forceinline__ void stage16B(const __bf16* g, __bf16* l) {
#ifdef HAVE_ASYNC_LDS
    // signature (from compiler diagnostic): (v4i AS(1)*, v4i AS(3)*, imm offset, imm cpol)
    __builtin_amdgcn_global_load_async_to_lds_b128(
        (__attribute__((address_space(1))) v4i*)g,
        (__attribute__((address_space(3))) v4i*)l,
        0, 0);
#else
    *(v8bf*)l = *(const v8bf*)g;
#endif
}

__device__ __forceinline__ v16bf frag16(const __bf16* p0, const __bf16* p1) {
    v8bf lo = *(const v8bf*)p0;
    v8bf hi = *(const v8bf*)p1;
    return __builtin_shufflevector(lo, hi, 0,1,2,3,4,5,6,7,8,9,10,11,12,13,14,15);
}

// =====================================================================
// Pass 1a: x (f32) -> x_bf16
// =====================================================================
__global__ __launch_bounds__(256) void conv_x_bf16(const float* __restrict__ x,
                                                   __bf16* __restrict__ xb) {
    const size_t i4 = (size_t)blockIdx.x * blockDim.x + threadIdx.x; // float4 group
    const size_t g  = i4 * 4;
    v4f v = *(const v4f*)(x + g);
    v4bf o;
    #pragma unroll
    for (int c = 0; c < 4; ++c) o[c] = (__bf16)v[c];
    *(v4bf*)(xb + g) = o;
}

// =====================================================================
// Pass 1b: W_bf16[s,o,i] = bf16(w_mu[o,i] + exp(w_ls[o,i]) * r1[s,o,i])
// =====================================================================
__global__ __launch_bounds__(256) void conv_w_bf16(const float* __restrict__ w_mu,
                                                   const float* __restrict__ w_ls,
                                                   const float* __restrict__ r1,
                                                   __bf16* __restrict__ wb) {
    const size_t i4 = (size_t)blockIdx.x * blockDim.x + threadIdx.x; // float4 group
    const size_t g  = i4 * 4;                                       // element index in [S*1M)
    const int    wi = (int)(g & (size_t)(N_OUT * N_IN - 1));        // 2^20 elements per sample
    v4f mu = *(const v4f*)(w_mu + wi);
    v4f ls = *(const v4f*)(w_ls + wi);
    v4f rr = *(const v4f*)(r1 + g);
    v4bf o;
    #pragma unroll
    for (int c = 0; c < 4; ++c) o[c] = (__bf16)(mu[c] + __expf(ls[c]) * rr[c]);
    *(v4bf*)(wb + g) = o;
}

// =====================================================================
// Pass 2: y[s] = x_bf16 @ W_bf16[s]^T + bias   (256x128 tile, async LDS,
//         double-buffered, pure WMMA inner loop)
// =====================================================================
#define GM 256   // batch-tile rows
#define GN 128   // out-tile cols

__global__ __launch_bounds__(256) void gemm_bf16_wmma(
    const __bf16* __restrict__ xb,   // [BATCH, N_IN] bf16
    const __bf16* __restrict__ wb,   // [S, N_OUT, N_IN] bf16
    const float*  __restrict__ b_mu,
    const float*  __restrict__ b_ls,
    const float*  __restrict__ r2,
    float*        __restrict__ y)    // [S, BATCH, N_OUT]
{
    __shared__ __bf16 sX[2][GM * LDSS];   // 2 x 20 KB
    __shared__ __bf16 sW[2][GN * LDSS];   // 2 x 10 KB

    const int s  = blockIdx.z;
    const int b0 = blockIdx.y * GM;
    const int o0 = blockIdx.x * GN;

    const int t    = threadIdx.x;
    const int lane = t & 31;
    const int wave = t >> 5;       // 0..7
    const int wm   = wave & 3;     // 4 waves along M (64 rows each)
    const int wn   = wave >> 2;    // 2 waves along N (64 cols each)
    const int h    = lane >> 4;
    const int l    = lane & 15;

    const __bf16* __restrict__ wsamp = wb + (size_t)s * N_OUT * N_IN;

    // Accumulators: 4 m-tiles x 4 n-tiles of 16x16 f32 (128 VGPRs)
    v8f acc[4][4];
    #pragma unroll
    for (int i = 0; i < 4; ++i)
        #pragma unroll
        for (int j = 0; j < 4; ++j)
            #pragma unroll
            for (int r = 0; r < 8; ++r)
                acc[i][j][r] = 0.0f;

    // Staging: 16B chunks. X tile: GM rows x 2 chunks = 512 -> 2/thread.
    //          W tile: GN rows x 2 chunks = 256 -> 1/thread.  (3 per thread)
    const int xr0 = t >> 1,          xc0 = (t & 1) * 16;
    const int xr1 = (t + 256) >> 1,  xc1 = ((t + 256) & 1) * 16;
    const int wr  = t >> 1,          wc  = (t & 1) * 16;

    #define ISSUE_TILE(k0, buf)                                                          \
        do {                                                                             \
            stage16B(xb + (size_t)(b0 + xr0) * N_IN + (k0) + xc0,                        \
                     &sX[buf][xr0 * LDSS + xc0]);                                        \
            stage16B(xb + (size_t)(b0 + xr1) * N_IN + (k0) + xc1,                        \
                     &sX[buf][xr1 * LDSS + xc1]);                                        \
            stage16B(wsamp + (size_t)(o0 + wr) * N_IN + (k0) + wc,                       \
                     &sW[buf][wr * LDSS + wc]);                                          \
        } while (0)

    ISSUE_TILE(0, 0);

    for (int k0 = 0, it = 0; k0 < N_IN; k0 += BK, ++it) {
        const int buf = it & 1;
        if (k0 + BK < N_IN) {
            ISSUE_TILE(k0 + BK, buf ^ 1);   // prefetch next chunk into other buffer
            ASYNC_WAIT(3);                  // current chunk's 3 copies complete (in-order)
        } else {
            ASYNC_WAIT(0);
        }
        __syncthreads();                    // all waves' chunk data visible in LDS

        // A fragments (16x32 bf16): lane(l,h): row=l, halves [h*8..h*8+7], [16+h*8..]
        v16bf afr[4];
        #pragma unroll
        for (int mt = 0; mt < 4; ++mt) {
            const int row = wm * 64 + mt * 16 + l;
            afr[mt] = frag16(&sX[buf][row * LDSS + h * 8],
                             &sX[buf][row * LDSS + 16 + h * 8]);
        }
        // B fragments (32x16 bf16): lane(l,h): N=l, halves [h*8..h*8+7], [16+h*8..]
        v16bf bfr[4];
        #pragma unroll
        for (int nt = 0; nt < 4; ++nt) {
            const int row = wn * 64 + nt * 16 + l;
            bfr[nt] = frag16(&sW[buf][row * LDSS + h * 8],
                             &sW[buf][row * LDSS + 16 + h * 8]);
        }

        // 16 WMMAs per wave per K-step
        #pragma unroll
        for (int mt = 0; mt < 4; ++mt)
            #pragma unroll
            for (int nt = 0; nt < 4; ++nt)
                acc[mt][nt] = __builtin_amdgcn_wmma_f32_16x16x32_bf16(
                    false, afr[mt], false, bfr[nt],
                    (short)0, acc[mt][nt], false, false);

        __syncthreads();                    // safe to refill this buffer next round
    }
    #undef ISSUE_TILE

    // Epilogue: bias + nontemporal streaming stores.
    // C/D layout: lanes 0-15 -> N=lane, M=r ; lanes 16-31 -> N=lane-16, M=8+r
    const size_t yoff = (size_t)s * BATCH * N_OUT;
    #pragma unroll
    for (int nt = 0; nt < 4; ++nt) {
        const int o = o0 + wn * 64 + nt * 16 + l;
        const float bias = b_mu[o] + __expf(b_ls[o]) * r2[(size_t)s * N_OUT + o];
        #pragma unroll
        for (int mt = 0; mt < 4; ++mt) {
            const int mbase = b0 + wm * 64 + mt * 16 + h * 8;
            #pragma unroll
            for (int r = 0; r < 8; ++r) {
                __builtin_nontemporal_store(acc[mt][nt][r] + bias,
                    y + yoff + (size_t)(mbase + r) * N_OUT + o);
            }
        }
    }
}

// =====================================================================
// Fallback: fused single-pass kernel (round-1, compiles & works without ws)
// =====================================================================
#define BM 128
#define BN 128

__global__ __launch_bounds__(256) void blinear_fused(
    const float* __restrict__ x, const float* __restrict__ w_mu,
    const float* __restrict__ w_ls, const float* __restrict__ b_mu,
    const float* __restrict__ b_ls, const float* __restrict__ r1,
    const float* __restrict__ r2, float* __restrict__ y)
{
    __shared__ __bf16 sX[BM * LDSS];
    __shared__ __bf16 sW[BN * LDSS];

    const int s  = blockIdx.z;
    const int b0 = blockIdx.y * BM;
    const int o0 = blockIdx.x * BN;

    const int t    = threadIdx.x;
    const int lane = t & 31;
    const int wave = t >> 5;
    const int wm   = wave & 1;
    const int wn   = wave >> 1;
    const int h    = lane >> 4;
    const int l    = lane & 15;

    const float* __restrict__ r1s = r1 + (size_t)s * N_OUT * N_IN;

    v8f acc[4][2];
    #pragma unroll
    for (int i = 0; i < 4; ++i)
        #pragma unroll
        for (int j = 0; j < 2; ++j)
            #pragma unroll
            for (int r = 0; r < 8; ++r)
                acc[i][j][r] = 0.0f;

    v4f px[4], pmu[4], pls[4], pr[4];
    #pragma unroll
    for (int i = 0; i < 4; ++i) {
        const int slot = t + i * 256;
        const int row  = slot >> 3;
        const int col  = (slot & 7) * 4;
        px[i]  = *(const v4f*)(x + (size_t)(b0 + row) * N_IN + col);
        const size_t g = (size_t)(o0 + row) * N_IN + col;
        pmu[i] = *(const v4f*)(w_mu + g);
        pls[i] = *(const v4f*)(w_ls + g);
        pr[i]  = *(const v4f*)(r1s + g);
    }

    for (int k0 = 0; k0 < N_IN; k0 += BK) {
        #pragma unroll
        for (int i = 0; i < 4; ++i) {
            const int slot = t + i * 256;
            const int row  = slot >> 3;
            const int col  = (slot & 7) * 4;
            v4bf xbv, wbv;
            #pragma unroll
            for (int c = 0; c < 4; ++c) {
                xbv[c] = (__bf16)px[i][c];
                wbv[c] = (__bf16)(pmu[i][c] + __expf(pls[i][c]) * pr[i][c]);
            }
            *(v4bf*)&sX[row * LDSS + col] = xbv;
            *(v4bf*)&sW[row * LDSS + col] = wbv;
        }
        __syncthreads();

        const int kn = k0 + BK;
        if (kn < N_IN) {
            #pragma unroll
            for (int i = 0; i < 4; ++i) {
                const int slot = t + i * 256;
                const int row  = slot >> 3;
                const int col  = (slot & 7) * 4;
                px[i]  = *(const v4f*)(x + (size_t)(b0 + row) * N_IN + kn + col);
                const size_t g = (size_t)(o0 + row) * N_IN + kn + col;
                pmu[i] = *(const v4f*)(w_mu + g);
                pls[i] = *(const v4f*)(w_ls + g);
                pr[i]  = *(const v4f*)(r1s + g);
            }
        }

        v16bf afr[4];
        #pragma unroll
        for (int mt = 0; mt < 4; ++mt) {
            const int row = wm * 64 + mt * 16 + l;
            afr[mt] = frag16(&sX[row * LDSS + h * 8], &sX[row * LDSS + 16 + h * 8]);
        }
        v16bf bfr[2];
        #pragma unroll
        for (int nt = 0; nt < 2; ++nt) {
            const int row = wn * 32 + nt * 16 + l;
            bfr[nt] = frag16(&sW[row * LDSS + h * 8], &sW[row * LDSS + 16 + h * 8]);
        }

        #pragma unroll
        for (int mt = 0; mt < 4; ++mt)
            #pragma unroll
            for (int nt = 0; nt < 2; ++nt)
                acc[mt][nt] = __builtin_amdgcn_wmma_f32_16x16x32_bf16(
                    false, afr[mt], false, bfr[nt],
                    (short)0, acc[mt][nt], false, false);

        __syncthreads();
    }

    const size_t yoff = (size_t)s * BATCH * N_OUT;
    #pragma unroll
    for (int nt = 0; nt < 2; ++nt) {
        const int o = o0 + wn * 32 + nt * 16 + l;
        const float bias = b_mu[o] + __expf(b_ls[o]) * r2[(size_t)s * N_OUT + o];
        #pragma unroll
        for (int mt = 0; mt < 4; ++mt) {
            const int mbase = b0 + wm * 64 + mt * 16 + h * 8;
            #pragma unroll
            for (int r = 0; r < 8; ++r) {
                __builtin_nontemporal_store(acc[mt][nt][r] + bias,
                    y + yoff + (size_t)(mbase + r) * N_OUT + o);
            }
        }
    }
}

// =====================================================================
// Host launcher
// =====================================================================
extern "C" void kernel_launch(void* const* d_in, const int* in_sizes, int n_in,
                              void* d_out, int out_size, void* d_ws, size_t ws_size,
                              hipStream_t stream) {
    const float* x    = (const float*)d_in[0];
    const float* w_mu = (const float*)d_in[1];
    const float* w_ls = (const float*)d_in[2];
    const float* b_mu = (const float*)d_in[3];
    const float* b_ls = (const float*)d_in[4];
    const float* r1   = (const float*)d_in[5];
    const float* r2   = (const float*)d_in[6];
    float* y = (float*)d_out;

    const int S = in_sizes[5] / (N_OUT * N_IN);   // 64

    const size_t w_bytes = (size_t)S * N_OUT * N_IN * 2;   // 128 MB bf16
    const size_t x_bytes = (size_t)BATCH * N_IN * 2;       // 8 MB bf16

    if (ws_size >= w_bytes + x_bytes) {
        __bf16* wb = (__bf16*)d_ws;
        __bf16* xb = (__bf16*)((char*)d_ws + w_bytes);

        // Pass 1: convert operands to bf16 (workspace fits in 192 MB L2)
        conv_x_bf16<<<(BATCH * N_IN) / (256 * 4), 256, 0, stream>>>(x, xb);
        conv_w_bf16<<<(unsigned)(((size_t)S * N_OUT * N_IN) / (256 * 4)), 256, 0, stream>>>(
            w_mu, w_ls, r1, wb);

        // Pass 2: 64 batched GEMMs, 256x128 tiles
        dim3 grid(N_OUT / GN, BATCH / GM, S);   // (8, 16, 64)
        gemm_bf16_wmma<<<grid, 256, 0, stream>>>(xb, wb, b_mu, b_ls, r2, y);
    } else {
        dim3 grid(N_OUT / BN, BATCH / BM, S);   // (8, 32, 64)
        blinear_fused<<<grid, 256, 0, stream>>>(x, w_mu, w_ls, b_mu, b_ls, r1, r2, y);
    }
}
